// TrackGNN_28406913696308
// MI455X (gfx1250) — matrix operand order repaired and checked by reading, not compiled, not run on hardware
//
#include <hip/hip_runtime.h>
#include <hip/hip_bf16.h>

typedef __attribute__((ext_vector_type(16))) _Float16 v16h;
typedef __attribute__((ext_vector_type(8)))  float    v8f;

// Problem constants (match reference)
#define BG_N   32      // nodes per graph
#define BG_E   31      // edges per graph
#define ND_    16
#define ED_    8
#define GD_    8
#define HN_    128
#define HE_    128
#define KN_PAD 64      // node MLP K (40) padded to 2x32
#define KE_PAD 288     // edge MLP K (272) padded to 9x32
#define KT_N   2       // node K tiles of 32
#define KT_E   9       // edge K tiles of 32
#define NT_    8       // N tiles of 16 (128 wide)

// Packed weight sizes (halves)
#define WN_PACK_HALVES (NT_ * KT_N * 32 * 16)   // 8192
#define WE_PACK_HALVES (NT_ * KT_E * 32 * 16)   // 36864
#define PACK_TOTAL     (WN_PACK_HALVES + WE_PACK_HALVES)

// ---------------------------------------------------------------------------
// Weight packing: fp32 [K,128] -> f16 B-fragments.
// Fragment layout (per WMMA B 32x16): lane&15 = N column, lane>>4 selects
// K base (0 or 16), 16 contiguous halves = ascending K. Stored so that the
// main kernel reads one contiguous v16h per lane:
//   idx = ((nt*KT + kt)*32 + lane)*16 + j
// ---------------------------------------------------------------------------
__global__ void pack_weights_kernel(const float* __restrict__ Wn,
                                    const float* __restrict__ We,
                                    _Float16* __restrict__ wnp,
                                    _Float16* __restrict__ wep) {
    int idx = blockIdx.x * blockDim.x + threadIdx.x;
    if (idx < WN_PACK_HALVES) {
        int j    = idx & 15;
        int lane = (idx >> 4) & 31;
        int kt   = (idx >> 9) & (KT_N - 1);
        int nt   = idx >> 10;
        int k = kt * 32 + (lane >> 4) * 16 + j;
        int n = nt * 16 + (lane & 15);
        wnp[idx] = (_Float16)((k < (ND_ + 2 * ED_ + GD_)) ? Wn[k * HN_ + n] : 0.0f);
    } else if (idx < PACK_TOTAL) {
        int t    = idx - WN_PACK_HALVES;
        int j    = t & 15;
        int lane = (t >> 4) & 31;
        int rem  = t >> 9;
        int kt   = rem % KT_E;
        int nt   = rem / KT_E;
        int k = kt * 32 + (lane >> 4) * 16 + j;
        int n = nt * 16 + (lane & 15);
        wep[t] = (_Float16)((k < (ED_ + 2 * HN_ + GD_)) ? We[k * HE_ + n] : 0.0f);
    }
}

// ---------------------------------------------------------------------------
// A-fragment loader (16x32 f16, contiguous K layout matching packed B):
//   lanes 0-15 : row M = mbase+lane,    K = kt*32 + {0..15}
//   lanes 16-31: row M = mbase+lane-16, K = kt*32 + {16..31}
// Loaded as two 16-byte LDS reads.
// ---------------------------------------------------------------------------
__device__ __forceinline__ v16h load_a_frag(const _Float16* A, int ld,
                                            int mbase, int kt, int lane) {
    int l  = lane & 15;
    int hi = lane >> 4;
    const _Float16* p = A + (mbase + l) * ld + kt * 32 + hi * 16;
    v16h a;
    *reinterpret_cast<uint4*>(&a)       = *reinterpret_cast<const uint4*>(p);
    *(reinterpret_cast<uint4*>(&a) + 1) = *reinterpret_cast<const uint4*>(p + 8);
    return a;
}

struct __align__(16) SMem {
    _Float16 node_in[BG_N][KN_PAD];   // 4 KB  node MLP input (f16, K-padded)
    _Float16 h_node[BG_N][HN_];       // 8 KB  node MLP output (f16)
    _Float16 edge_in[BG_N][KE_PAD];   // 18 KB edge MLP input (f16, K-padded)
    float    h_edge[BG_N][HE_];       // 16 KB edge MLP output (f32)
    float    ef[BG_N][ED_];           // raw edge features (row 31 = 0)
    float    emask[BG_N];             // edge mask (31 valid)
    float    gvec[GD_];
    float    spart[BG_N][8];          // score partial sums
    float    score[BG_N];
    float    wgt[BG_N];
    float    red[2];                  // [0]=max, [1]=1/sum
    float    agg[HE_];
    float    hglob[HN_];
    float    hcls[64];
};

__global__ void __launch_bounds__(256)
trackgnn_fused_kernel(const float* __restrict__ GF,   // [B,8]
                      const float* __restrict__ NF,   // [B,32,16]
                      const float* __restrict__ EF,   // [B,31,8]
                      const int*   __restrict__ NM,   // [B,32]
                      const _Float16* __restrict__ WnP,
                      const _Float16* __restrict__ WeP,
                      const float* __restrict__ BN,   // [128]
                      const float* __restrict__ BE,   // [128]
                      const float* __restrict__ WS,   // [128]
                      const float* __restrict__ BS,   // [1]
                      const float* __restrict__ WG,   // [128,128]
                      const float* __restrict__ BGv,  // [128]
                      const float* __restrict__ WC1,  // [128,64]
                      const float* __restrict__ BC1,  // [64]
                      const float* __restrict__ WC2,  // [64]
                      const float* __restrict__ BC2,  // [1]
                      float* __restrict__ Y,          // [B]
                      int B) {
    __shared__ SMem sm;

    const int tid  = threadIdx.x;
    const int w    = tid >> 5;        // wave id 0..7 -> owns N tile w
    const int lane = tid & 31;
    const int l    = lane & 15;
    const int hi   = lane >> 4;

    // ---- per-wave resident state: B fragments + biases (reused all graphs)
    const v16h* wnv = reinterpret_cast<const v16h*>(WnP);
    const v16h* wev = reinterpret_cast<const v16h*>(WeP);
    v16h wn_frag[KT_N];
    v16h we_frag[KT_E];
#pragma unroll
    for (int kt = 0; kt < KT_N; ++kt) wn_frag[kt] = wnv[(w * KT_N + kt) * 32 + lane];
#pragma unroll
    for (int kt = 0; kt < KT_E; ++kt) we_frag[kt] = wev[(w * KT_E + kt) * 32 + lane];
    const float biasn = BN[w * 16 + l];
    const float biase = BE[w * 16 + l];
    const float bs0  = BS[0];
    const float bc20 = BC2[0];

    for (int g = blockIdx.x; g < B; g += gridDim.x) {
        // ------------------ Phase A: stage inputs, build node_input -------
        if (tid < GD_) sm.gvec[tid] = GF[g * GD_ + tid];
        if (tid < BG_N) {
            float em = 0.0f;
            if (tid < BG_E) {
                int m0 = NM[g * BG_N + tid];
                int m1 = NM[g * BG_N + tid + 1];
                em = (m0 != 0 && m1 != 0) ? 1.0f : 0.0f;
            }
            sm.emask[tid] = em;
        }
        {
            int row = tid >> 3, c = tid & 7;
            sm.ef[row][c] = (row < BG_E) ? EF[g * (BG_E * ED_) + row * ED_ + c] : 0.0f;
        }
        __syncthreads();
        {
            int row = tid >> 3, seg = tid & 7;
            _Float16* dst = &sm.node_in[row][seg * 8];
            if (seg < 2) {                       // node features
                const float* p = NF + g * (BG_N * ND_) + row * ND_ + seg * 8;
#pragma unroll
                for (int c = 0; c < 8; ++c) dst[c] = (_Float16)p[c];
            } else if (seg == 2) {               // edge_backward = ef_masked[row-1]
                if (row > 0) {
                    float em = sm.emask[row - 1];
#pragma unroll
                    for (int c = 0; c < 8; ++c) dst[c] = (_Float16)(sm.ef[row - 1][c] * em);
                } else {
#pragma unroll
                    for (int c = 0; c < 8; ++c) dst[c] = (_Float16)0.0f;
                }
            } else if (seg == 3) {               // edge_forward = ef_masked[row]
                float em = (row < BG_E) ? sm.emask[row] : 0.0f;
#pragma unroll
                for (int c = 0; c < 8; ++c)
                    dst[c] = (_Float16)((row < BG_E) ? sm.ef[row][c] * em : 0.0f);
            } else if (seg == 4) {               // global features
#pragma unroll
                for (int c = 0; c < 8; ++c) dst[c] = (_Float16)sm.gvec[c];
            } else {                             // K padding 40..63
#pragma unroll
                for (int c = 0; c < 8; ++c) dst[c] = (_Float16)0.0f;
            }
        }
        __syncthreads();

        // ------------------ Phase B: node GEMM (WMMA) + ReLU --------------
        // A-fragment double-buffer: issue kt+1's LDS loads before consuming
        // kt's fragment so s_wait_dscnt can retire early.
#pragma unroll
        for (int mt = 0; mt < 2; ++mt) {
            v8f acc;
#pragma unroll
            for (int r = 0; r < 8; ++r) acc[r] = 0.0f;
            v16h a_cur = load_a_frag(&sm.node_in[0][0], KN_PAD, mt * 16, 0, lane);
#pragma unroll
            for (int kt = 0; kt < KT_N; ++kt) {
                v16h a_nxt;
                if (kt + 1 < KT_N)
                    a_nxt = load_a_frag(&sm.node_in[0][0], KN_PAD, mt * 16, kt + 1, lane);
                acc = __builtin_amdgcn_wmma_f32_16x16x32_f16(
                    false, a_cur, false, wn_frag[kt], (short)0, acc, false, false);
                if (kt + 1 < KT_N) a_cur = a_nxt;
            }
#pragma unroll
            for (int r = 0; r < 8; ++r) {
                float v = acc[r] + biasn;
                v = v > 0.0f ? v : 0.0f;
                sm.h_node[mt * 16 + hi * 8 + r][w * 16 + l] = (_Float16)v;
            }
        }
        __syncthreads();

        // ------------------ Phase C: assemble edge_input ------------------
        // row e: [ef(8) | h_node[e](128) | h_node[e+1](128) | g(8) | pad(16)]
        // all region boundaries are multiples of 8 halves (16B aligned).
        for (int idx = tid; idx < BG_N * (KE_PAD / 8); idx += 256) {
            int row = idx / (KE_PAD / 8);
            int ch  = idx - row * (KE_PAD / 8);
            _Float16* dst = &sm.edge_in[row][ch * 8];
            if (row >= BG_E || ch >= 34) {
#pragma unroll
                for (int c = 0; c < 8; ++c) dst[c] = (_Float16)0.0f;
            } else if (ch == 0) {
#pragma unroll
                for (int c = 0; c < 8; ++c) dst[c] = (_Float16)sm.ef[row][c];
            } else if (ch <= 16) {
                *reinterpret_cast<uint4*>(dst) =
                    *reinterpret_cast<const uint4*>(&sm.h_node[row][(ch - 1) * 8]);
            } else if (ch <= 32) {
                *reinterpret_cast<uint4*>(dst) =
                    *reinterpret_cast<const uint4*>(&sm.h_node[row + 1][(ch - 17) * 8]);
            } else { // ch == 33
#pragma unroll
                for (int c = 0; c < 8; ++c) dst[c] = (_Float16)sm.gvec[c];
            }
        }
        __syncthreads();

        // ------------------ Phase D: edge GEMM (WMMA) + ReLU --------------
        // Two independent accumulator chains (even/odd kt) + A double-buffer:
        // hides WMMA D->C latency and LDS latency inside the wave.
#pragma unroll
        for (int mt = 0; mt < 2; ++mt) {
            v8f acc0, acc1;
#pragma unroll
            for (int r = 0; r < 8; ++r) { acc0[r] = 0.0f; acc1[r] = 0.0f; }
            v16h a0 = load_a_frag(&sm.edge_in[0][0], KE_PAD, mt * 16, 0, lane);
            v16h a1 = load_a_frag(&sm.edge_in[0][0], KE_PAD, mt * 16, 1, lane);
#pragma unroll
            for (int kt = 0; kt < KT_E; kt += 2) {
                v16h n0, n1;
                if (kt + 2 < KT_E)
                    n0 = load_a_frag(&sm.edge_in[0][0], KE_PAD, mt * 16, kt + 2, lane);
                if (kt + 3 < KT_E)
                    n1 = load_a_frag(&sm.edge_in[0][0], KE_PAD, mt * 16, kt + 3, lane);
                acc0 = __builtin_amdgcn_wmma_f32_16x16x32_f16(
                    false, a0, false, we_frag[kt], (short)0, acc0, false, false);
                if (kt + 1 < KT_E)
                    acc1 = __builtin_amdgcn_wmma_f32_16x16x32_f16(
                        false, a1, false, we_frag[kt + 1], (short)0, acc1, false, false);
                if (kt + 2 < KT_E) a0 = n0;
                if (kt + 3 < KT_E) a1 = n1;
            }
#pragma unroll
            for (int r = 0; r < 8; ++r) {
                float v = acc0[r] + acc1[r] + biase;
                v = v > 0.0f ? v : 0.0f;
                sm.h_edge[mt * 16 + hi * 8 + r][w * 16 + l] = v;
            }
        }
        __syncthreads();

        // ------------------ Phase E: masked softmax pooling ---------------
        {
            int e = tid >> 3, jj = tid & 7;
            if (e < BG_E) {
                float p = 0.0f;
                const float* he = sm.h_edge[e];
#pragma unroll
                for (int c = 0; c < 16; ++c) p += he[jj * 16 + c] * WS[jj * 16 + c];
                sm.spart[e][jj] = p;
            }
        }
        __syncthreads();
        if (tid < BG_E) {
            float s = bs0;
#pragma unroll
            for (int jj = 0; jj < 8; ++jj) s += sm.spart[tid][jj];
            float em = sm.emask[tid];
            s = s * em + (1.0f - em) * -1e9f;
            sm.score[tid] = s;
        }
        __syncthreads();
        if (tid == 0) {
            float mx = -3.4e38f;
            for (int e = 0; e < BG_E; ++e) mx = fmaxf(mx, sm.score[e]);
            sm.red[0] = mx;
        }
        __syncthreads();
        if (tid < BG_E) sm.wgt[tid] = __expf(sm.score[tid] - sm.red[0]);
        __syncthreads();
        if (tid == 0) {
            float s = 0.0f;
            for (int e = 0; e < BG_E; ++e) s += sm.wgt[e];
            sm.red[1] = 1.0f / s;
        }
        __syncthreads();
        if (tid < HE_) {
            float a = 0.0f;
            for (int e = 0; e < BG_E; ++e) a += sm.h_edge[e][tid] * sm.wgt[e];
            sm.agg[tid] = a * sm.red[1];
        }
        __syncthreads();

        // ------------------ Phase F: global update + classifier -----------
        if (tid < HN_) {
            float a = BGv[tid];
            for (int i = 0; i < HE_; ++i) a += sm.agg[i] * WG[i * HN_ + tid];
            sm.hglob[tid] = fmaxf(a, 0.0f);
        }
        __syncthreads();
        if (tid < 64) {
            float a = BC1[tid];
            for (int i = 0; i < HN_; ++i) a += sm.hglob[i] * WC1[i * 64 + tid];
            sm.hcls[tid] = fmaxf(a, 0.0f);
        }
        __syncthreads();
        if (tid == 0) {
            float y = bc20;
            for (int o = 0; o < 64; ++o) y += sm.hcls[o] * WC2[o];
            Y[g] = y;
        }
        __syncthreads();   // protect LDS reuse across graph-loop iterations
    }
}

extern "C" void kernel_launch(void* const* d_in, const int* in_sizes, int n_in,
                              void* d_out, int out_size, void* d_ws, size_t ws_size,
                              hipStream_t stream) {
    const float* GF  = (const float*)d_in[0];
    const float* NF  = (const float*)d_in[1];
    const float* EF  = (const float*)d_in[2];
    const int*   NM  = (const int*)d_in[3];
    const float* Wn  = (const float*)d_in[4];
    const float* BN  = (const float*)d_in[5];
    const float* We  = (const float*)d_in[6];
    const float* BE  = (const float*)d_in[7];
    const float* WS  = (const float*)d_in[8];
    const float* BS  = (const float*)d_in[9];
    const float* WG  = (const float*)d_in[10];
    const float* BGv = (const float*)d_in[11];
    const float* WC1 = (const float*)d_in[12];
    const float* BC1 = (const float*)d_in[13];
    const float* WC2 = (const float*)d_in[14];
    const float* BC2 = (const float*)d_in[15];

    _Float16* wnp = (_Float16*)d_ws;
    _Float16* wep = wnp + WN_PACK_HALVES;

    const int B = in_sizes[0] / GD_;   // 8192

    pack_weights_kernel<<<(PACK_TOTAL + 255) / 256, 256, 0, stream>>>(Wn, We, wnp, wep);

    int grid = 1024;
    if (grid > B) grid = B;
    trackgnn_fused_kernel<<<grid, 256, 0, stream>>>(
        GF, NF, EF, NM, wnp, wep, BN, BE, WS, BS, WG, BGv, WC1, BC1, WC2, BC2,
        (float*)d_out, B);
}
